// Sc_Dot_Pro_Attention_73478300500427
// MI455X (gfx1250) — matrix operand order repaired and checked by reading, not compiled, not run on hardware
//
#include <hip/hip_runtime.h>

typedef __bf16 bf16;
typedef unsigned int u32;
typedef __attribute__((ext_vector_type(16))) __bf16 v16bf;
typedef __attribute__((ext_vector_type(8)))  float  v8f;
typedef __attribute__((ext_vector_type(4)))  u32    u32x4;
typedef __attribute__((ext_vector_type(4)))  float  f32x4;
typedef int gi32x4 __attribute__((vector_size(16)));   // matches async-LDS builtin param

union BFrag { v16bf v; u32x4 u[2]; };

constexpr int S_LEN = 4096;   // sequence length
constexpr int EMB   = 512;    // embedding dim
constexpr int HD    = 64;     // head dim
constexpr int NB    = 4;      // batch

// gfx1250 async global->LDS path (ASYNCcnt), with synchronous fallback.
#if defined(__has_builtin)
#  if __has_builtin(__builtin_amdgcn_global_load_async_to_lds_b128) && \
      __has_builtin(__builtin_amdgcn_s_wait_asynccnt)
#    define HAVE_ASYNC_LDS 1
#  endif
#endif
#ifndef HAVE_ASYNC_LDS
#  define HAVE_ASYNC_LDS 0
#endif

#define GLOBAL_AS __attribute__((address_space(1)))
#define LDS_AS    __attribute__((address_space(3)))

// ---------------------------------------------------------------------------
// Kernel 0: transpose W[512][64] (x3) into bf16 Wt[3][64][512] in workspace.
// ---------------------------------------------------------------------------
__global__ __launch_bounds__(256) void wprep_kernel(
    const float* __restrict__ Wq, const float* __restrict__ Wk,
    const float* __restrict__ Wv, bf16* __restrict__ Wt) {
  const int total = 3 * HD * EMB;
  for (int idx = blockIdx.x * blockDim.x + threadIdx.x; idx < total;
       idx += gridDim.x * blockDim.x) {
    int p   = idx >> 15;          // / (64*512)
    int rem = idx & 32767;
    int n   = rem >> 9;           // / 512
    int k   = rem & 511;
    const float* W = (p == 0) ? Wq : ((p == 1) ? Wk : Wv);
    Wt[idx] = (bf16)W[k * HD + n];
  }
}

// ---------------------------------------------------------------------------
// Kernel 1: fused QKV projection.  One wave computes a 16-row tile of all
// three 64-col projections via v_wmma_f32_16x16x32_bf16 (12 accumulators).
// Q gets the 1/sqrt(64)=0.125 softmax scale folded in.  V is stored
// transposed per batch: Vt[b][64][4096].
// ---------------------------------------------------------------------------
__global__ __launch_bounds__(256) void qkv_kernel(
    const float* __restrict__ X, const bf16* __restrict__ Wt,
    const float* __restrict__ bq, const float* __restrict__ bk,
    const float* __restrict__ bv,
    bf16* __restrict__ Qg, bf16* __restrict__ Kg, bf16* __restrict__ Vt) {
  const int tid  = threadIdx.x;
  const int wave = tid >> 5;
  const int lane = tid & 31;
  const int nl   = lane & 15;
  const int half = lane >> 4;
  const int kb   = half * 8;                 // A-fragment K sub-offset
  const int m0   = blockIdx.x * 128 + wave * 16;

  const v8f vzero = {0.f, 0.f, 0.f, 0.f, 0.f, 0.f, 0.f, 0.f};
  v8f acc[3][4];
#pragma unroll
  for (int p = 0; p < 3; ++p)
#pragma unroll
    for (int j = 0; j < 4; ++j) acc[p][j] = vzero;

  const float* xr = X + (size_t)(m0 + nl) * EMB + kb;

  for (int k0 = 0; k0 < EMB; k0 += 32) {
    // A fragment: X rows m0..m0+15, K = k0..k0+31 (bf16 convert on the fly)
    f32x4 a0 = *(const f32x4*)(xr + k0);
    f32x4 a1 = *(const f32x4*)(xr + k0 + 4);
    f32x4 a2 = *(const f32x4*)(xr + k0 + 16);
    f32x4 a3 = *(const f32x4*)(xr + k0 + 20);
    v16bf af;
    af[0]  = (bf16)a0.x; af[1]  = (bf16)a0.y; af[2]  = (bf16)a0.z; af[3]  = (bf16)a0.w;
    af[4]  = (bf16)a1.x; af[5]  = (bf16)a1.y; af[6]  = (bf16)a1.z; af[7]  = (bf16)a1.w;
    af[8]  = (bf16)a2.x; af[9]  = (bf16)a2.y; af[10] = (bf16)a2.z; af[11] = (bf16)a2.w;
    af[12] = (bf16)a3.x; af[13] = (bf16)a3.y; af[14] = (bf16)a3.z; af[15] = (bf16)a3.w;

#pragma unroll
    for (int p = 0; p < 3; ++p) {
#pragma unroll
      for (int j = 0; j < 4; ++j) {
        const bf16* wb =
            Wt + (size_t)(p * 64 + 16 * j + nl) * EMB + k0 + half * 8;
        BFrag wf;
        wf.u[0] = *(const u32x4*)(wb);
        wf.u[1] = *(const u32x4*)(wb + 16);
        acc[p][j] = __builtin_amdgcn_wmma_f32_16x16x32_bf16(
            false, af, false, wf.v, (short)0, acc[p][j], false, false);
      }
    }
  }

  // Epilogue: bias add, bf16 convert, store (Q scaled by 0.125, V transposed)
#pragma unroll
  for (int j = 0; j < 4; ++j) {
    const int col = 16 * j + nl;
    const float bjq = bq[col], bjk = bk[col], bjv = bv[col];
#pragma unroll
    for (int r = 0; r < 8; ++r) {
      const int g = m0 + r + 8 * half;       // global row in [0, B*S)
      Qg[(size_t)g * HD + col] = (bf16)((acc[0][j][r] + bjq) * 0.125f);
      Kg[(size_t)g * HD + col] = (bf16)(acc[1][j][r] + bjk);
      const int bb = g >> 12;                // / 4096
      const int ss = g & (S_LEN - 1);
      Vt[((size_t)bb * HD + col) * S_LEN + ss] = (bf16)(acc[2][j][r] + bjv);
    }
  }
}

// ---------------------------------------------------------------------------
// Kernel 2: flash attention.  blockDim=128 (4 waves); each block owns 64
// query rows of one batch, each wave a 16-row WMMA tile.  Loops over 64
// KV tiles of 64 with DOUBLE-BUFFERED LDS: tile t+1 is fetched with
// gfx1250 async global->LDS B128 loads (ASYNCcnt) while tile t runs
// through the WMMA + online-softmax pipeline.  One barrier per tile.
// ---------------------------------------------------------------------------
__global__ __launch_bounds__(128) void attn_kernel(
    const bf16* __restrict__ Qg, const bf16* __restrict__ Kg,
    const bf16* __restrict__ Vt, float* __restrict__ out) {
  __shared__ __align__(16) bf16 Klds[2][64 * 64];   // K tiles, row-major [kv][d]
  __shared__ __align__(16) bf16 Vlds[2][64 * 64];   // V^T tiles [d][kv]
  __shared__ __align__(16) bf16 Plds[4 * 16 * 64];  // per-wave P scratch

  const int b    = blockIdx.y;
  const int tid  = threadIdx.x;
  const int wave = tid >> 5;
  const int lane = tid & 31;
  const int nl   = lane & 15;
  const int half = lane >> 4;
  const int kb   = half * 8;

  const int q0 = blockIdx.x * 64 + wave * 16;       // query row within batch
  const bf16* Qb = Qg + (size_t)b * S_LEN * HD;
  const bf16* Kb = Kg + (size_t)b * S_LEN * HD;
  const bf16* Vb = Vt + (size_t)b * HD * S_LEN;

  // Issue the staging loads for KV tile t into buffer (t&1).
  auto issue_tile = [&](int t) {
    const int kv0 = t * 64;
    bf16* kd = Klds[t & 1];
    bf16* vd = Vlds[t & 1];
#pragma unroll
    for (int i = 0; i < 4; ++i) {
      const int vidx = i * 128 + tid;
      const int rr = vidx >> 3, cc = vidx & 7;
      const bf16* ks = Kb + (size_t)(kv0 + rr) * HD + cc * 8;
      const bf16* vs = Vb + (size_t)rr * S_LEN + kv0 + cc * 8;
#if HAVE_ASYNC_LDS
      __builtin_amdgcn_global_load_async_to_lds_b128(
          (GLOBAL_AS gi32x4*)ks, (LDS_AS gi32x4*)(kd + (size_t)vidx * 8), 0, 0);
      __builtin_amdgcn_global_load_async_to_lds_b128(
          (GLOBAL_AS gi32x4*)vs, (LDS_AS gi32x4*)(vd + (size_t)vidx * 8), 0, 0);
#else
      ((u32x4*)kd)[vidx] = *(const u32x4*)ks;
      ((u32x4*)vd)[vidx] = *(const u32x4*)vs;
#endif
    }
  };

  // Q A-fragments (2 K-chunks of 32), kept in registers for the whole kernel
  BFrag qf[2];
  {
    const bf16* qr = Qb + (size_t)(q0 + nl) * HD + kb;
    qf[0].u[0] = *(const u32x4*)(qr);
    qf[0].u[1] = *(const u32x4*)(qr + 16);
    qf[1].u[0] = *(const u32x4*)(qr + 32);
    qf[1].u[1] = *(const u32x4*)(qr + 48);
  }

  const v8f vzero = {0.f, 0.f, 0.f, 0.f, 0.f, 0.f, 0.f, 0.f};
  float m_i[8], l_i[8];
  v8f o[4];
#pragma unroll
  for (int r = 0; r < 8; ++r) { m_i[r] = -__builtin_inff(); l_i[r] = 0.f; }
#pragma unroll
  for (int j = 0; j < 4; ++j) o[j] = vzero;

  issue_tile(0);  // prologue: prime buffer 0

  for (int t = 0; t < 64; ++t) {
    // tile t's async loads (issued last iteration) must have landed
#if HAVE_ASYNC_LDS
    __builtin_amdgcn_s_wait_asynccnt(0);
#endif
    __syncthreads();
    if (t + 1 < 64) issue_tile(t + 1);  // overlap next fetch with compute

    const bf16* kt = Klds[t & 1];
    const bf16* vt = Vlds[t & 1];

    // --- scores S = Q K^T (softmax scale pre-folded into Q) ---
    v8f s[4];
#pragma unroll
    for (int j = 0; j < 4; ++j) s[j] = vzero;
#pragma unroll
    for (int c = 0; c < 2; ++c) {
#pragma unroll
      for (int j = 0; j < 4; ++j) {
        const bf16* kp = kt + (16 * j + nl) * 64 + 32 * c + half * 8;
        BFrag kf;
        kf.u[0] = *(const u32x4*)(kp);
        kf.u[1] = *(const u32x4*)(kp + 16);
        s[j] = __builtin_amdgcn_wmma_f32_16x16x32_bf16(
            false, qf[c].v, false, kf.v, (short)0, s[j], false, false);
      }
    }

    // --- online softmax (reductions within 16-lane halves = N dimension) ---
    float rm[8];
#pragma unroll
    for (int r = 0; r < 8; ++r)
      rm[r] = fmaxf(fmaxf(s[0][r], s[1][r]), fmaxf(s[2][r], s[3][r]));
#pragma unroll
    for (int mask = 1; mask <= 8; mask <<= 1)
#pragma unroll
      for (int r = 0; r < 8; ++r)
        rm[r] = fmaxf(rm[r], __shfl_xor(rm[r], mask, 32));

    float alpha[8];
#pragma unroll
    for (int r = 0; r < 8; ++r) {
      const float mn = fmaxf(m_i[r], rm[r]);
      alpha[r] = __expf(m_i[r] - mn);
      m_i[r] = mn;
    }
#pragma unroll
    for (int j = 0; j < 4; ++j)
#pragma unroll
      for (int r = 0; r < 8; ++r)
        s[j][r] = __expf(s[j][r] - m_i[r]);

    float rs[8];
#pragma unroll
    for (int r = 0; r < 8; ++r)
      rs[r] = (s[0][r] + s[1][r]) + (s[2][r] + s[3][r]);
#pragma unroll
    for (int mask = 1; mask <= 8; mask <<= 1)
#pragma unroll
      for (int r = 0; r < 8; ++r)
        rs[r] += __shfl_xor(rs[r], mask, 32);
#pragma unroll
    for (int r = 0; r < 8; ++r)
      l_i[r] = l_i[r] * alpha[r] + rs[r];
#pragma unroll
    for (int j = 0; j < 4; ++j)
#pragma unroll
      for (int r = 0; r < 8; ++r)
        o[j][r] *= alpha[r];

    // --- P: D-layout -> A-layout via per-wave LDS bounce ---
    bf16* pw = Plds + wave * (16 * 64);
#pragma unroll
    for (int j = 0; j < 4; ++j)
#pragma unroll
      for (int r = 0; r < 8; ++r)
        pw[(r + 8 * half) * 64 + 16 * j + nl] = (bf16)s[j][r];

    __asm__ volatile("s_wait_dscnt 0" ::: "memory");

    BFrag pf[2];
    {
      const bf16* pr = pw + nl * 64 + kb;
      pf[0].u[0] = *(const u32x4*)(pr);
      pf[0].u[1] = *(const u32x4*)(pr + 16);
      pf[1].u[0] = *(const u32x4*)(pr + 32);
      pf[1].u[1] = *(const u32x4*)(pr + 48);
    }

    // --- O += P V ---
#pragma unroll
    for (int c = 0; c < 2; ++c) {
#pragma unroll
      for (int j = 0; j < 4; ++j) {
        const bf16* vp = vt + (16 * j + nl) * 64 + 32 * c + half * 8;
        BFrag vf;
        vf.u[0] = *(const u32x4*)(vp);
        vf.u[1] = *(const u32x4*)(vp + 16);
        o[j] = __builtin_amdgcn_wmma_f32_16x16x32_bf16(
            false, pf[c].v, false, vf.v, (short)0, o[j], false, false);
      }
    }
  }

  // --- final normalize + fp32 store ---
#pragma unroll
  for (int j = 0; j < 4; ++j)
#pragma unroll
    for (int r = 0; r < 8; ++r) {
      const int srow = q0 + r + 8 * half;
      out[((size_t)(b * S_LEN + srow)) * HD + 16 * j + nl] = o[j][r] / l_i[r];
    }
}

// ---------------------------------------------------------------------------
extern "C" void kernel_launch(void* const* d_in, const int* in_sizes, int n_in,
                              void* d_out, int out_size, void* d_ws, size_t ws_size,
                              hipStream_t stream) {
  (void)in_sizes; (void)n_in; (void)out_size; (void)ws_size;
  const float* X  = (const float*)d_in[0];
  const float* Wq = (const float*)d_in[1];
  const float* bq = (const float*)d_in[2];
  const float* Wk = (const float*)d_in[3];
  const float* bk = (const float*)d_in[4];
  const float* Wv = (const float*)d_in[5];
  const float* bv = (const float*)d_in[6];

  // workspace layout (bf16 elements): Wt | Q | K | V^T   (~6.5 MB total)
  bf16* Wt = (bf16*)d_ws;
  bf16* Qg = Wt + (size_t)3 * HD * EMB;
  bf16* Kg = Qg + (size_t)NB * S_LEN * HD;
  bf16* Vt = Kg + (size_t)NB * S_LEN * HD;

  wprep_kernel<<<384, 256, 0, stream>>>(Wq, Wk, Wv, Wt);
  qkv_kernel<<<128, 256, 0, stream>>>(X, Wt, bq, bk, bv, Qg, Kg, Vt);
  attn_kernel<<<dim3(64, NB), 128, 0, stream>>>(Qg, Kg, Vt, (float*)d_out);
}
